// EdgeLoss3D_38354057953739
// MI455X (gfx1250) — matrix operand order repaired and checked by reading, not compiled, not run on hardware
//
#include <hip/hip_runtime.h>
#include <hip/hip_bf16.h>

typedef __attribute__((ext_vector_type(16))) _Float16 v16h;
typedef __attribute__((ext_vector_type(8)))  float    v8f;

#define DIMD 128
#define DIMH 192
#define DIMW 192
#define NB   4
#define NS   18            // loaded D slices (16 outputs + 2 halo)
#define TILE 16            // in-plane tile incl. halo
#define VAL  14            // valid interior outputs per tile dim
#define NTH  ((DIMH + VAL - 1) / VAL)   // 14
#define NTW  ((DIMW + VAL - 1) / VAL)   // 14

__global__ void edge_zero_kernel(float* out) {
    if (threadIdx.x == 0) out[0] = 0.0f;
}

__global__ void __launch_bounds__(256)
edge_loss3d_kernel(const float* __restrict__ pred,
                   const float* __restrict__ tgt,
                   float* __restrict__ out)
{
    __shared__ _Float16 sX[NS][TILE][TILE];        // input tile (f16)
    __shared__ _Float16 sZ[8][2][TILE][TILE];      // per-wave Z^T scratch
    __shared__ _Float16 sT[3][NS][TILE][TILE];     // in-plane filtered fields
    __shared__ float    sE[16 * VAL * VAL];        // edge magnitude of pred
    __shared__ float    sSum[256];

    const int tid  = threadIdx.x;
    const int wave = tid >> 5;
    const int lane = tid & 31;
    const int n    = lane & 15;
    const int hi   = (lane >= 16) ? 1 : 0;

    const int b  = blockIdx.z;
    const int d0 = blockIdx.y * 16;
    const int th = blockIdx.x / NTW;
    const int tw = blockIdx.x % NTW;
    const int h0 = th * VAL;
    const int w0 = tw * VAL;

    // ---- loop-invariant banded coefficient matrices (WMMA f16 operands) ----
    // B-layout (32x16): lane -> N=lane%16, half t -> K=(t<8?t:t+8)+(lane<16?0:8)
    // aSs/aDs (16x32), contiguous K order matching bzs/bzd: lane -> M=lane%16, half t -> K=t+(lane<16?0:16)
    v16h bSw = {}, bDw = {}, aSs = {}, aDs = {};
    {
        const int kb = hi ? 8 : 0;
        for (int t = 0; t < 16; ++t) {
            int k = ((t < 8) ? t : t + 8) + kb;
            int d = k - n;
            float sm = (d == 0) ? 0.5f : ((d == 1 || d == -1) ? 0.25f : 0.0f);
            float sb = (d == 1) ? 1.0f : ((d == -1) ? -1.0f : 0.0f);
            if (k >= 16) { sm = 0.0f; sb = 0.0f; }   // K-padding region
            bSw[t] = (_Float16)sm;
            bDw[t] = (_Float16)sb;
        }
        const int ab = hi ? 16 : 0;
        for (int t = 0; t < 16; ++t) {
            int k = t + ab;
            int d = n - k;                            // m - K with m = lane%16
            float sm = (d == 0) ? 0.5f : ((d == 1 || d == -1) ? 0.25f : 0.0f);
            float sb = (d == 1) ? 1.0f : ((d == -1) ? -1.0f : 0.0f);
            if (k >= 16) { sm = 0.0f; sb = 0.0f; }
            aSs[t] = (_Float16)sm;
            aDs[t] = (_Float16)sb;
        }
    }

    float lsum = 0.0f;

    for (int pass = 0; pass < 2; ++pass) {
        const float* __restrict__ src = pass ? tgt : pred;
        __syncthreads();

        // ---- global -> LDS (f32 -> f16), zero padding outside the volume ----
        for (int e = tid; e < NS * TILE * TILE; e += 256) {
            int s = e >> 8, r = (e >> 4) & 15, c = e & 15;
            int gd = d0 - 1 + s, gh = h0 - 1 + r, gw = w0 - 1 + c;
            float v = 0.0f;
            if (gd >= 0 && gd < DIMD && gh >= 0 && gh < DIMH && gw >= 0 && gw < DIMW)
                v = src[(((long long)b * DIMD + gd) * DIMH + gh) * (long long)DIMW + gw];
            sX[s][r][c] = (_Float16)v;
        }
        __syncthreads();

        // ---- per-D-slice separable in-plane conv on the matrix pipe ----
        for (int s = wave; s < NS; s += 8) {
            // A = data rows (K = W column; K>=16 zero-padded)
            v16h a = {};
            {
                const _Float16* xr = &sX[s][n][hi ? 8 : 0];
                for (int t = 0; t < 8; ++t) a[t] = xr[t];
            }
            // W-direction: Zs = X*Sw^T , Zd = X*Dw^T
            v8f zs = {}, zd = {};
            zs = __builtin_amdgcn_wmma_f32_16x16x32_f16(false, a, false, bSw,
                                                        (short)0, zs, false, false);
            zd = __builtin_amdgcn_wmma_f32_16x16x32_f16(false, a, false, bDw,
                                                        (short)0, zd, false, false);

            // Store Z transposed (f16) into per-wave scratch; D layout:
            // VGPR v holds M = v + (lane<16?0:8), N = lane%16
            const int mo = hi ? 8 : 0;
            for (int v = 0; v < 8; ++v) {
                sZ[wave][0][n][v + mo] = (_Float16)zs[v];
                sZ[wave][1][n][v + mo] = (_Float16)zd[v];
            }
            // B = Z rows (K = Z row index; lanes>=16 carry K>=16 -> zero)
            v16h bzs = {}, bzd = {};
            if (!hi) {
                for (int t = 0; t < 16; ++t) {
                    bzs[t] = sZ[wave][0][n][t];
                    bzd[t] = sZ[wave][1][n][t];
                }
            }
            // H-direction: t1 = Ss*Zs, t2 = Ss*Zd, t3 = Ds*Zs
            v8f t1 = {}, t2 = {}, t3 = {};
            t1 = __builtin_amdgcn_wmma_f32_16x16x32_f16(false, aSs, false, bzs,
                                                        (short)0, t1, false, false);
            t2 = __builtin_amdgcn_wmma_f32_16x16x32_f16(false, aSs, false, bzd,
                                                        (short)0, t2, false, false);
            t3 = __builtin_amdgcn_wmma_f32_16x16x32_f16(false, aDs, false, bzs,
                                                        (short)0, t3, false, false);
            for (int v = 0; v < 8; ++v) {
                int m = v + mo;
                sT[0][s][m][n] = (_Float16)t1[v];   // smoothH(smoothW)  -> gz path
                sT[1][s][m][n] = (_Float16)t2[v];   // smoothH(sobelW)   -> gx path
                sT[2][s][m][n] = (_Float16)t3[v];   // sobelH(smoothW)   -> gy path
            }
        }
        __syncthreads();

        // ---- D-direction pass + gradient magnitude (VALU) ----
        for (int idx = tid; idx < 16 * VAL * VAL; idx += 256) {
            int j  = idx / (VAL * VAL);
            int r  = idx % (VAL * VAL);
            int i  = r / VAL, wv = r % VAL;
            int hh = i + 1, ww = wv + 1;
            float gx = 0.25f * (float)sT[1][j][hh][ww]
                     + 0.5f  * (float)sT[1][j + 1][hh][ww]
                     + 0.25f * (float)sT[1][j + 2][hh][ww];
            float gy = 0.25f * (float)sT[2][j][hh][ww]
                     + 0.5f  * (float)sT[2][j + 1][hh][ww]
                     + 0.25f * (float)sT[2][j + 2][hh][ww];
            float gz = (float)sT[0][j + 2][hh][ww] - (float)sT[0][j][hh][ww];
            float e  = sqrtf(gx * gx + gy * gy + gz * gz + 1e-8f);
            if (pass == 0) {
                sE[idx] = e;
            } else {
                if ((h0 + i) < DIMH && (w0 + wv) < DIMW)
                    lsum += fabsf(sE[idx] - e);
            }
        }
    }

    // ---- block reduction + single global accumulate ----
    sSum[tid] = lsum;
    __syncthreads();
    for (int s = 128; s > 0; s >>= 1) {
        if (tid < s) sSum[tid] += sSum[tid + s];
        __syncthreads();
    }
    if (tid == 0) {
        const float invN = 1.0f / (float)((long long)NB * DIMD * DIMH * DIMW);
        atomicAdd(out, sSum[0] * invN);
    }
}

extern "C" void kernel_launch(void* const* d_in, const int* in_sizes, int n_in,
                              void* d_out, int out_size, void* d_ws, size_t ws_size,
                              hipStream_t stream) {
    (void)in_sizes; (void)n_in; (void)out_size; (void)d_ws; (void)ws_size;
    const float* pred = (const float*)d_in[0];
    const float* tgt  = (const float*)d_in[1];
    float* out = (float*)d_out;

    edge_zero_kernel<<<dim3(1), dim3(64), 0, stream>>>(out);
    dim3 grid(NTH * NTW, DIMD / 16, NB);
    edge_loss3d_kernel<<<grid, dim3(256), 0, stream>>>(pred, tgt, out);
}
